// LlamaAttention_36928128811665
// MI455X (gfx1250) — matrix operand order repaired and checked by reading, not compiled, not run on hardware
//
#include <hip/hip_runtime.h>
#include <hip/hip_bf16.h>
#include <cstdint>

// ---------------------------------------------------------------------------
// Llama attention on gfx1250 (MI455X): bf16 WMMA + async global->LDS staging.
// B=4, S=1024, H=4096, NH=32, NKV=8, D=128, THETA=10000
// ---------------------------------------------------------------------------

#define DEV __device__ __forceinline__

typedef __bf16 bf16;
typedef bf16  v16bf  __attribute__((ext_vector_type(16)));
typedef bf16  bf16x2 __attribute__((ext_vector_type(2)));
typedef float v8f    __attribute__((ext_vector_type(8)));

static constexpr int Bn    = 4;
static constexpr int Sn    = 1024;
static constexpr int Hn    = 4096;
static constexpr int NH    = 32;
static constexpr int NKV   = 8;
static constexpr int Dh    = 128;
static constexpr int QKVC  = (NH + 2 * NKV) * Dh;   // 6144
static constexpr int Mrows = Bn * Sn;               // 4096
static constexpr float SCALE = 0.08838834764831845f; // 1/sqrt(128)

// A-matrix (16x32, 16-bit) K-pair base for vgpr v, half hi (ISA 7.12.2)
DEV int kpA(int v, int hi) { return (v < 4 ? 2 * v : 16 + 2 * (v - 4)) + hi * 8; }
// B-matrix (32x16, 16-bit) K-pair base, same k order as kpA
DEV int kpB(int v, int hi) { return (v < 4 ? 2 * v : 16 + 2 * (v - 4)) + hi * 8; }

DEV v16bf frag_a(const bf16* rowptr, int koff, int hi) {
    v16bf f;
#pragma unroll
    for (int v = 0; v < 8; ++v) {
        int kp = kpA(v, hi) + koff;
        f[2 * v]     = rowptr[kp];
        f[2 * v + 1] = rowptr[kp + 1];
    }
    return f;
}

DEV v16bf frag_b(const bf16* rowptr, int koff, int hi) {
    v16bf f;
#pragma unroll
    for (int v = 0; v < 8; ++v) {
        int kp = kpB(v, hi) + koff;
        f[2 * v]     = rowptr[kp];
        f[2 * v + 1] = rowptr[kp + 1];
    }
    return f;
}

DEV v8f wmma_bf16(v16bf a, v16bf b, v8f c) {
    return __builtin_amdgcn_wmma_f32_16x16x32_bf16(false, a, false, b,
                                                   (short)0, c, false, false);
}

// async global -> LDS copy of 16 bytes per lane; tracked by ASYNCcnt
DEV void async_copy_b128(unsigned lds_addr, unsigned long long gaddr) {
    asm volatile("global_load_async_to_lds_b128 %0, %1, off"
                 :: "v"(lds_addr), "v"(gaddr) : "memory");
}
DEV unsigned lds_off(const void* p) { return (unsigned)(uintptr_t)p; }

// ---------------------------------------------------------------------------
// fp32 -> bf16 convert (straight) and convert+transpose (for weights)
// ---------------------------------------------------------------------------
__global__ __launch_bounds__(256)
void convert_f32_bf16(const float* __restrict__ in, bf16* __restrict__ out) {
    size_t i = (size_t)blockIdx.x * 256 + threadIdx.x;
    out[i] = (bf16)in[i];
}

// in[R][C] fp32 -> out[C][R] bf16
__global__ __launch_bounds__(256)
void transpose_f32_bf16(const float* __restrict__ in, bf16* __restrict__ out,
                        int R, int C) {
    size_t i = (size_t)blockIdx.x * 256 + threadIdx.x;
    int r = (int)(i / C);
    int c = (int)(i % C);
    out[(size_t)c * R + r] = (bf16)in[i];
}

// ---------------------------------------------------------------------------
// GEMM: C[M,N] = A[M,K] * BT[N,K]^T, A/BT bf16, f32 accumulate, C = TC.
// 256 threads (8 waves); block tile 128x256; wave tile 64x64; K-tile 32.
// 16 WMMAs per wave per K-step against 8 fragment loads (1 ds_load_b128
// per WMMA). LDS double-buffered, filled by global_load_async_to_lds_b128.
// ---------------------------------------------------------------------------
template <typename TC>
__global__ __launch_bounds__(256)
void gemm_async_wmma(const bf16* __restrict__ A, const bf16* __restrict__ BT,
                     TC* __restrict__ C, int M, int N, int K) {
    // row stride 40 bf16 = 80 bytes (16B aligned, staggers banks)
    __shared__ bf16 As[2][128][40];
    __shared__ bf16 Bs[2][256][40];

    const int t     = threadIdx.x;
    const int lane  = t & 31;
    const int wid   = t >> 5;
    const int waveM = wid >> 2;          // 0..1 -> 64 rows
    const int waveN = wid & 3;           // 0..3 -> 64 cols
    const int hi    = lane >> 4;
    const int lm    = lane & 15;

    const int m0 = blockIdx.y * 128;
    const int n0 = blockIdx.x * 256;
    const int nTiles = K >> 5;

    // per thread per K-tile: 2 chunks of A + 4 chunks of B (16B each)
    auto issue_tile = [&](int kt, int buf) {
#pragma unroll
        for (int j = 0; j < 2; ++j) {
            int id = t + j * 256;        // 0..511
            int r  = id >> 2;            // 0..127
            int c  = id & 3;             // 16B chunk in the 64B row segment
            unsigned la = lds_off(&As[buf][r][0]) + c * 16;
            unsigned long long ga =
                (unsigned long long)(uintptr_t)(A + (size_t)(m0 + r) * K + kt * 32) + c * 16;
            async_copy_b128(la, ga);
        }
#pragma unroll
        for (int j = 0; j < 4; ++j) {
            int id = t + j * 256;        // 0..1023
            int r  = id >> 2;            // 0..255
            int c  = id & 3;
            unsigned lb = lds_off(&Bs[buf][r][0]) + c * 16;
            unsigned long long gb =
                (unsigned long long)(uintptr_t)(BT + (size_t)(n0 + r) * K + kt * 32) + c * 16;
            async_copy_b128(lb, gb);
        }
    };

    v8f acc[4][4] = {};

    issue_tile(0, 0);
    for (int kt = 0; kt < nTiles; ++kt) {
        int buf = kt & 1;
        if (kt + 1 < nTiles) {
            issue_tile(kt + 1, buf ^ 1);
            // asyncs complete in order: <=6 outstanding => tile kt has landed
            asm volatile("s_wait_asynccnt 0x6" ::: "memory");
        } else {
            asm volatile("s_wait_asynccnt 0x0" ::: "memory");
        }
        __syncthreads();   // all waves' copies for buf are visible

        v16bf bfr[4];
#pragma unroll
        for (int nt = 0; nt < 4; ++nt)
            bfr[nt] = frag_b(&Bs[buf][waveN * 64 + nt * 16 + lm][0], 0, hi);
#pragma unroll
        for (int mt = 0; mt < 4; ++mt) {
            v16bf af = frag_a(&As[buf][waveM * 64 + mt * 16 + lm][0], 0, hi);
#pragma unroll
            for (int nt = 0; nt < 4; ++nt)
                acc[mt][nt] = wmma_bf16(af, bfr[nt], acc[mt][nt]);
        }
        __syncthreads();   // done reading buf before kt+2's asyncs overwrite it
    }

    // C layout: vgpr v, lanes0-15 -> M=v, lanes16-31 -> M=v+8; N = lane&15
#pragma unroll
    for (int mt = 0; mt < 4; ++mt)
#pragma unroll
        for (int nt = 0; nt < 4; ++nt) {
            int row0 = m0 + waveM * 64 + mt * 16 + hi * 8;
            int col  = n0 + waveN * 64 + nt * 16 + lm;
            TC* cp = C + (size_t)row0 * N + col;
#pragma unroll
            for (int v = 0; v < 8; ++v)
                cp[(size_t)v * N] = (TC)acc[mt][nt][v];
        }
}

// ---------------------------------------------------------------------------
// RoPE in place on bf16 q (heads 0..31) and k (heads 32..39) in qkv buffer.
// ---------------------------------------------------------------------------
__global__ __launch_bounds__(256)
void rope_kernel(const int* __restrict__ positions, bf16* __restrict__ qkv) {
    size_t idx = (size_t)blockIdx.x * 256 + threadIdx.x;
    int i = (int)(idx & 63);
    idx >>= 6;
    int head = (int)(idx % (NH + NKV));   // 0..39
    size_t row = idx / (NH + NKV);        // 0..4095
    if (row >= (size_t)Mrows) return;

    float pos = (float)positions[row];
    float inv = __powf(10000.0f, -(float)i * (1.0f / 64.0f));
    float sn, cs;
    __sincosf(pos * inv, &sn, &cs);

    size_t base = row * QKVC + (size_t)head * Dh;   // q heads then k heads contiguous
    float x1 = (float)qkv[base + i];
    float x2 = (float)qkv[base + i + 64];
    qkv[base + i]      = (bf16)(x1 * cs - x2 * sn);
    qkv[base + i + 64] = (bf16)(x2 * cs + x1 * sn);
}

// ---------------------------------------------------------------------------
// Flash attention, causal. Grid: (B*NH, S/64). Block: 128 threads (4 waves).
// K tile staged via async-to-LDS; V staged with a VALU transpose; bf16 WMMA.
// ---------------------------------------------------------------------------
__global__ __launch_bounds__(128)
void flash_attn(const bf16* __restrict__ qkv, bf16* __restrict__ attn) {
    __shared__ bf16 Ks[64][136];   // [key][d], 272B row stride (16B aligned)
    __shared__ bf16 Vs[128][72];   // [d][key] transposed
    __shared__ bf16 Ps[4][16][64]; // per-wave P scratch

    const int bh   = blockIdx.x;
    const int b    = bh / NH;
    const int h    = bh % NH;
    const int hkv  = h / (NH / NKV);
    const int qt   = blockIdx.y;
    const int wave = threadIdx.x >> 5;
    const int lane = threadIdx.x & 31;
    const int hi   = lane >> 4;
    const int lm   = lane & 15;
    const int q0   = qt * 64 + wave * 16;

    // preload Q fragments (4 chunks of K=32 over D=128) from global bf16
    v16bf qf[4];
    {
        const bf16* qp = qkv + (size_t)(b * Sn + q0 + lm) * QKVC + (size_t)h * Dh;
#pragma unroll
        for (int c = 0; c < 4; ++c)
#pragma unroll
            for (int v = 0; v < 8; ++v) {
                int kp = kpA(v, hi) + c * 32;
                bf16x2 p = *(const bf16x2*)&qp[kp];
                qf[c][2 * v]     = p.x;
                qf[c][2 * v + 1] = p.y;
            }
    }

    v8f   o[8] = {};
    float m_i[8], l_i[8];
#pragma unroll
    for (int v = 0; v < 8; ++v) { m_i[v] = -3.0e38f; l_i[v] = 0.0f; }

    for (int kb = 0; kb <= qt; ++kb) {
        __syncthreads();  // previous iteration done reading Ks/Vs

        // K tile: 64 keys x 256B rows = 1024 16B chunks; async copy
        {
            size_t kbase = (size_t)(b * Sn + kb * 64) * QKVC + (size_t)NH * Dh
                         + (size_t)hkv * Dh;
#pragma unroll
            for (int j = 0; j < 8; ++j) {
                int id  = threadIdx.x + j * 128;   // 0..1023
                int key = id >> 4;
                int c   = id & 15;
                unsigned la = lds_off(&Ks[key][0]) + c * 16;
                unsigned long long ga =
                    (unsigned long long)(uintptr_t)(qkv + kbase + (size_t)key * QKVC) + c * 16;
                async_copy_b128(la, ga);
            }
        }
        // V tile: transpose to [d][key] through VALU
        {
            size_t vbase = (size_t)(b * Sn + kb * 64) * QKVC
                         + (size_t)(NH + NKV) * Dh + (size_t)hkv * Dh;
            for (int idx = threadIdx.x; idx < 64 * 128; idx += 128) {
                int key = idx >> 7;
                int d   = idx & 127;
                Vs[d][key] = qkv[vbase + (size_t)key * QKVC + d];
            }
        }
        asm volatile("s_wait_asynccnt 0x0" ::: "memory");
        __syncthreads();

        // scores S = Q K^T : 4 d-chunks x 4 key-tiles.
        // Load the 4 K-fragments of a chunk into distinct registers first so
        // the ds_load_b128s overlap instead of draining DScnt per WMMA.
        v8f sacc[4] = {};
#pragma unroll
        for (int c = 0; c < 4; ++c) {
            v16bf kf[4];
#pragma unroll
            for (int nt = 0; nt < 4; ++nt)
                kf[nt] = frag_b(&Ks[nt * 16 + lm][0], c * 32, hi);
#pragma unroll
            for (int nt = 0; nt < 4; ++nt)
                sacc[nt] = wmma_bf16(qf[c], kf[nt], sacc[nt]);
        }

        // scale + causal mask (diagonal key block only)
#pragma unroll
        for (int nt = 0; nt < 4; ++nt)
#pragma unroll
            for (int v = 0; v < 8; ++v) {
                float s = sacc[nt][v] * SCALE;
                if (kb == qt) {
                    int qg = q0 + v + hi * 8;
                    int kg = kb * 64 + nt * 16 + lm;
                    if (kg > qg) s = -3.0e38f;
                }
                sacc[nt][v] = s;
            }

        // online softmax (row stats replicated across 16-lane halves)
#pragma unroll
        for (int v = 0; v < 8; ++v) {
            float rm = sacc[0][v];
#pragma unroll
            for (int nt = 1; nt < 4; ++nt) rm = fmaxf(rm, sacc[nt][v]);
            rm = fmaxf(rm, __shfl_xor(rm, 1));
            rm = fmaxf(rm, __shfl_xor(rm, 2));
            rm = fmaxf(rm, __shfl_xor(rm, 4));
            rm = fmaxf(rm, __shfl_xor(rm, 8));
            float mN    = fmaxf(m_i[v], rm);
            float alpha = __expf(m_i[v] - mN);
            m_i[v] = mN;
            float rs = 0.0f;
#pragma unroll
            for (int nt = 0; nt < 4; ++nt) {
                float p = __expf(sacc[nt][v] - mN);
                sacc[nt][v] = p;
                rs += p;
            }
            rs += __shfl_xor(rs, 1);
            rs += __shfl_xor(rs, 2);
            rs += __shfl_xor(rs, 4);
            rs += __shfl_xor(rs, 8);
            l_i[v] = l_i[v] * alpha + rs;
#pragma unroll
            for (int nt = 0; nt < 8; ++nt) o[nt][v] *= alpha;
        }

        // P (C-layout) -> LDS row-major for reuse as an A-fragment
#pragma unroll
        for (int nt = 0; nt < 4; ++nt)
#pragma unroll
            for (int v = 0; v < 8; ++v)
                Ps[wave][v + hi * 8][nt * 16 + lm] = (bf16)sacc[nt][v];
        __syncthreads();

        // O += P V : 2 key-chunks x 8 d-tiles, V-fragments in groups of 4
#pragma unroll
        for (int c = 0; c < 2; ++c) {
            v16bf pf = frag_a(&Ps[wave][lm][0], c * 32, hi);
#pragma unroll
            for (int g = 0; g < 2; ++g) {
                v16bf vf[4];
#pragma unroll
                for (int j = 0; j < 4; ++j)
                    vf[j] = frag_b(&Vs[(g * 4 + j) * 16 + lm][0], c * 32, hi);
#pragma unroll
                for (int j = 0; j < 4; ++j)
                    o[g * 4 + j] = wmma_bf16(pf, vf[j], o[g * 4 + j]);
            }
        }
    }

    // finalize: divide by l, write attn[B,S,NH*D] as bf16 (feeds GEMM2)
#pragma unroll
    for (int v = 0; v < 8; ++v) {
        float invl = 1.0f / l_i[v];
        int row = b * Sn + q0 + v + hi * 8;
#pragma unroll
        for (int nt = 0; nt < 8; ++nt) {
            int col = h * Dh + nt * 16 + lm;
            attn[(size_t)row * (NH * Dh) + col] = (bf16)(o[nt][v] * invl);
        }
    }
}

// ---------------------------------------------------------------------------
// Launch
// ---------------------------------------------------------------------------
extern "C" void kernel_launch(void* const* d_in, const int* in_sizes, int n_in,
                              void* d_out, int out_size, void* d_ws, size_t ws_size,
                              hipStream_t stream) {
    const int*   positions = (const int*)d_in[0];
    const float* hidden    = (const float*)d_in[1];
    const float* w_qkv     = (const float*)d_in[2];
    const float* w_o       = (const float*)d_in[3];
    float*       out       = (float*)d_out;

    const size_t n_hid   = (size_t)Mrows * Hn;       // 16.8M
    const size_t n_qkvw  = (size_t)Hn * QKVC;        // 25.2M
    const size_t n_ow    = (size_t)Hn * Hn;          // 16.8M
    const size_t n_qkv   = (size_t)Mrows * QKVC;     // 25.2M
    const size_t n_attn  = (size_t)Mrows * (NH * Dh);// 16.8M
    const size_t total_bf = n_hid + n_qkvw + n_ow + n_qkv + n_attn;
    if (ws_size < total_bf * sizeof(bf16)) return;

    bf16* hidden_bf = (bf16*)d_ws;
    bf16* wqkvT_bf  = hidden_bf + n_hid;   // [6144][4096]
    bf16* woT_bf    = wqkvT_bf + n_qkvw;   // [4096][4096]
    bf16* qkv_bf    = woT_bf + n_ow;       // [4096][6144]
    bf16* attn_bf   = qkv_bf + n_qkv;      // [4096][4096]

    // 0) one-time bf16 conversion / weight transposes (bandwidth pass)
    convert_f32_bf16<<<(unsigned)(n_hid / 256), 256, 0, stream>>>(hidden, hidden_bf);
    transpose_f32_bf16<<<(unsigned)(n_qkvw / 256), 256, 0, stream>>>(w_qkv, wqkvT_bf, Hn, QKVC);
    transpose_f32_bf16<<<(unsigned)(n_ow / 256), 256, 0, stream>>>(w_o, woT_bf, Hn, Hn);

    // 1) QKV projection (async-fed WMMA), bf16 output
    gemm_async_wmma<bf16><<<dim3(QKVC / 256, Mrows / 128), 256, 0, stream>>>(
        hidden_bf, wqkvT_bf, qkv_bf, Mrows, QKVC, Hn);

    // 2) RoPE on q and k in place (bf16)
    {
        size_t total = (size_t)Mrows * (NH + NKV) * 64;
        rope_kernel<<<(unsigned)(total / 256), 256, 0, stream>>>(positions, qkv_bf);
    }

    // 3) causal flash attention -> bf16 attn
    flash_attn<<<dim3(Bn * NH, Sn / 64), 128, 0, stream>>>(qkv_bf, attn_bf);

    // 4) output projection (async-fed WMMA), fp32 output
    gemm_async_wmma<float><<<dim3(Hn / 256, Mrows / 128), 256, 0, stream>>>(
        attn_bf, woT_bf, out, Mrows, Hn, Hn);
}
